// SoftmaxAttention_69965017252057
// MI455X (gfx1250) — matrix operand order repaired and checked by reading, not compiled, not run on hardware
//
#include <hip/hip_runtime.h>
#include <hip/hip_bf16.h>

typedef _Float16 v16h __attribute__((ext_vector_type(16)));
typedef _Float16 v8h  __attribute__((ext_vector_type(8)));
typedef float    v8f  __attribute__((ext_vector_type(8)));

#define WMMA_F16(A,B,C) __builtin_amdgcn_wmma_f32_16x16x32_f16(false,(A),false,(B),(short)0,(C),false,false)

#define BB 4
#define HH 16
#define SS 2048
#define DD 64
#define LDK 72        // padded LDS row stride in halves (144B -> conflict-free banks)
#define NT (SS/64)    // 32 key tiles
#define TILESZ (64*DD)
#define LOG2E 1.44269504088896340736f

__device__ __forceinline__ v16h cat8(v8h a, v8h b){
  return __builtin_shufflevector(a,b,0,1,2,3,4,5,6,7,8,9,10,11,12,13,14,15);
}

// ---- DPP16 row-rotate butterfly max (pure VALU, no LDS) ----
template<int CTRL>
__device__ __forceinline__ float dpp_f32(float x){
  return __int_as_float(__builtin_amdgcn_mov_dpp(__float_as_int(x), CTRL, 0xf, 0xf, false));
}
__device__ __forceinline__ float row_max16(float x){
  x = fmaxf(x, dpp_f32<0x128>(x));  // row_ror:8
  x = fmaxf(x, dpp_f32<0x124>(x));  // row_ror:4
  x = fmaxf(x, dpp_f32<0x122>(x));  // row_ror:2
  x = fmaxf(x, dpp_f32<0x121>(x));  // row_ror:1
  return x;
}

// ---- tile prefetch helpers (registers <-> global / LDS) ----
__device__ __forceinline__ void ldK(float (&r)[16], const float* p){
#pragma unroll
  for (int j=0;j<16;j++) r[j] = p[j];
}
__device__ __forceinline__ void ldV(float (&r)[16], const float* p){
#pragma unroll
  for (int j=0;j<16;j++) r[j] = p[(size_t)j*DD];
}
__device__ __forceinline__ void stTile(_Float16* kdst, _Float16* vdst,
                                       const float (&kr)[16], const float (&vr)[16]){
  v8h a,b,c,d;
#pragma unroll
  for (int j=0;j<8;j++){
    a[j]=(_Float16)kr[j];  b[j]=(_Float16)kr[8+j];
    c[j]=(_Float16)vr[j];  d[j]=(_Float16)vr[8+j];
  }
  *(v8h*)kdst = a; *(v8h*)(kdst+8) = b;
  *(v8h*)vdst = c; *(v8h*)(vdst+8) = d;
}

__global__ __launch_bounds__(256)
void fa_fwd_kernel(const float* __restrict__ Q, const float* __restrict__ K,
                   const float* __restrict__ V, const float* __restrict__ Msk,
                   float* __restrict__ O)
{
  __shared__ __align__(16) _Float16 kf[2][64*LDK];   // K tile [key][d], double-buffered
  __shared__ __align__(16) _Float16 vt[2][64*LDK];   // V tile transposed [d][key], double-buffered
  __shared__ __align__(16) _Float16 pbuf[8*16*LDK];  // per-wave scratch (P f16 / epilogue f32)

  const int tiles = SS/128;
  const int bh = blockIdx.x / tiles;
  const int mt = blockIdx.x % tiles;
  const int bb = bh / HH;
  const float* qp = Q + (size_t)bh*SS*DD;
  const float* kp = K + (size_t)bh*SS*DD;
  const float* vp = V + (size_t)bh*SS*DD;
  const float* mp = Msk + (size_t)bb*SS;

  const int tid  = threadIdx.x;
  const int wave = tid >> 5;
  const int lane = tid & 31;
  const int n    = lane & 15;
  const int hi   = lane >> 4;
  const int m0   = mt*128 + wave*16;

  // per-thread staging addresses (constant across tiles)
  const int   krow = tid >> 2, kcol = (tid & 3)*16;
  const float* pk0 = kp + (size_t)krow*DD + kcol;
  const int   kdoff = krow*LDK + kcol;
  const int   vcol = tid & 63, vrow0 = (tid >> 6)*16;
  const float* pv0 = vp + (size_t)vrow0*DD + vcol;
  const int   vdoff = vcol*LDK + vrow0;

  // ---- Q A-fragments, scaled into the exp2 domain: (1/8)*log2(e) ----
  const float qs = 0.125f * LOG2E;
  v16h qa0, qa1;
  {
    const float* qr = qp + (size_t)(m0 + n)*DD + hi*8;
#pragma unroll
    for (int j=0;j<8;j++){
      qa0[j]   = (_Float16)(qr[j]    * qs);
      qa0[j+8] = (_Float16)(qr[16+j] * qs);
      qa1[j]   = (_Float16)(qr[32+j] * qs);
      qa1[j+8] = (_Float16)(qr[48+j] * qs);
    }
  }

  // all-ones B fragment for row-sum-by-WMMA
  v16h ones;
#pragma unroll
  for (int j=0;j<16;j++) ones[j] = (_Float16)1.0f;

  v8f o[4], ls;
#pragma unroll
  for (int d=0; d<4; d++)
#pragma unroll
    for (int r=0; r<8; r++) o[d][r] = 0.0f;
#pragma unroll
  for (int r=0;r<8;r++) ls[r] = 0.0f;

  float rmax[8];
#pragma unroll
  for (int r=0;r<8;r++) rmax[r] = -__builtin_inff();

  _Float16* pb = &pbuf[wave*16*LDK];

  // ---- pipeline prologue: tile 0 -> LDS buf0, tile 1 -> registers ----
  float krg[16], vrg[16];
  ldK(krg, pk0); ldV(vrg, pv0);
  stTile(&kf[0][kdoff], &vt[0][vdoff], krg, vrg);
  ldK(krg, pk0 + TILESZ); ldV(vrg, pv0 + TILESZ);

  for (int kt = 0; kt < NT; ++kt){
    const int cur = kt & 1;
    const int kbase = kt*64;
    __syncthreads();   // buf[cur] visible to all; buf[1-cur] free for reuse

    // overlap with compute: stage tile kt+1 into alternate buffer, prefetch kt+2
    if (kt+1 < NT) stTile(&kf[1-cur][kdoff], &vt[1-cur][vdoff], krg, vrg);
    if (kt+2 < NT){ ldK(krg, pk0 + (size_t)(kt+2)*TILESZ); ldV(vrg, pv0 + (size_t)(kt+2)*TILESZ); }

    const _Float16* kbuf = kf[cur];
    const _Float16* vbuf = vt[cur];

    // ---- scores S' = log2e * (Q @ K^T)/8  (+ scaled additive mask) ----
    v8f sacc[4];
#pragma unroll
    for (int nt=0; nt<4; nt++){
      const _Float16* kr = &kbuf[(nt*16 + n)*LDK + hi*8];
      v16h kb0 = cat8(*(const v8h*)kr,      *(const v8h*)(kr+16));  // d 0..31
      v16h kb1 = cat8(*(const v8h*)(kr+32), *(const v8h*)(kr+48));  // d 32..63
      v8f s8;
#pragma unroll
      for (int r=0;r<8;r++) s8[r] = 0.0f;
      s8 = WMMA_F16(qa0, kb0, s8);
      s8 = WMMA_F16(qa1, kb1, s8);
      const float mv   = mp[kbase + nt*16 + n];
      const float madd = -1000.0f*LOG2E*(1.0f - mv);
#pragma unroll
      for (int r=0;r<8;r++) s8[r] += madd;
      sacc[nt] = s8;
    }

    // ---- online softmax: max via DPP butterfly; exp2 native ----
    float corr[8];
#pragma unroll
    for (int r=0;r<8;r++){
      float mx = fmaxf(fmaxf(sacc[0][r],sacc[1][r]), fmaxf(sacc[2][r],sacc[3][r]));
      mx = row_max16(mx);
      const float nm = fmaxf(rmax[r], mx);
      corr[r] = __builtin_amdgcn_exp2f(rmax[r] - nm);
      rmax[r] = nm;
#pragma unroll
      for (int nt=0; nt<4; nt++)
        sacc[nt][r] = __builtin_amdgcn_exp2f(sacc[nt][r] - nm);
    }

    // ---- P: C-layout -> A-layout via per-wave LDS scratch ----
#pragma unroll
    for (int nt=0; nt<4; nt++)
#pragma unroll
      for (int r=0; r<8; r++)
        pb[(r + 8*hi)*LDK + nt*16 + n] = (_Float16)sacc[nt][r];
    asm volatile("s_wait_dscnt 0" ::: "memory");  // wave-local DS ordering

    v16h pa0, pa1;
    {
      const _Float16* pr = &pb[n*LDK + hi*8];
      pa0 = cat8(*(const v8h*)pr,      *(const v8h*)(pr+16));   // keys 0..31
      pa1 = cat8(*(const v8h*)(pr+32), *(const v8h*)(pr+48));   // keys 32..63
    }

    // ---- rescale accumulators, then O += P@V and lsum += P@1 (all WMMA) ----
#pragma unroll
    for (int d=0; d<4; d++)
#pragma unroll
      for (int r=0;r<8;r++) o[d][r] *= corr[r];
#pragma unroll
    for (int r=0;r<8;r++) ls[r] *= corr[r];

    ls = WMMA_F16(pa0, ones, ls);
    ls = WMMA_F16(pa1, ones, ls);

#pragma unroll
    for (int d=0; d<4; d++){
      const _Float16* vr = &vbuf[(d*16 + n)*LDK + hi*8];
      v16h vb0 = cat8(*(const v8h*)vr,      *(const v8h*)(vr+16));  // keys 0..31
      v16h vb1 = cat8(*(const v8h*)(vr+32), *(const v8h*)(vr+48));  // keys 32..63
      o[d] = WMMA_F16(pa0, vb0, o[d]);
      o[d] = WMMA_F16(pa1, vb1, o[d]);
    }
  }

  // ---- epilogue: normalize, transpose via wave-local LDS, b128 stores ----
  float* ob = O + (size_t)bh*SS*DD;
  float* ew = (float*)pb;   // 2304B per wave, reuse P scratch (wave-local)
#pragma unroll
  for (int p=0; p<2; p++){
#pragma unroll
    for (int r=0; r<4; r++){
      const int rr = 4*p + r;
      const float inv = 1.0f / ls[rr];
      const int rowp = r + 4*hi;            // compacted row 0..7
#pragma unroll
      for (int d=0; d<4; d++)
        ew[rowp*64 + d*16 + n] = o[d][rr]*inv;
    }
    asm volatile("s_wait_dscnt 0" ::: "memory");
#pragma unroll
    for (int j=0; j<4; j++){
      const int c    = lane + 32*j;         // 0..127 chunk id
      const int rowp = c >> 4;              // 0..7
      const int col  = (c & 15)*4;
      float4 vv = *(float4*)&ew[rowp*64 + col];
      const int grow = m0 + 4*p + (rowp & 3) + 8*(rowp >> 2);
      *(float4*)(ob + (size_t)grow*DD + col) = vv;
    }
  }
}

extern "C" void kernel_launch(void* const* d_in, const int* in_sizes, int n_in,
                              void* d_out, int out_size, void* d_ws, size_t ws_size,
                              hipStream_t stream) {
  (void)in_sizes; (void)n_in; (void)out_size; (void)d_ws; (void)ws_size;
  const float* q = (const float*)d_in[0];
  const float* k = (const float*)d_in[1];
  const float* v = (const float*)d_in[2];
  const float* m = (const float*)d_in[3];
  float* out = (float*)d_out;
  dim3 grid(BB*HH*(SS/128));   // 1024 blocks
  fa_fwd_kernel<<<grid, 256, 0, stream>>>(q, k, v, m, out);
}
